// SelfMatchEncoder_43087111913646
// MI455X (gfx1250) — compile-verified, matrix operand order checked by reading, not run on hardware
//
#include <hip/hip_runtime.h>
#include <hip/hip_bf16.h>
#include <math.h>

// ---------------- problem constants ----------------
#define N_SEQ 1000
#define BATCH 8
#define DIM   256
#define HID   128
#define NPAD  1008   // query dim padded to multiple of 16
#define KPAD  1024   // doc dim padded to multiple of 32

// workspace layout (bytes)
#define VP_BYTES   (8000u * 128u * 4u)          //  4,096,000  vp   f32 [N*B][H]
#define P_ELEMS    (8u * 1008u * 1024u)         //  probs bf16 [B][NPAD][KPAD]
#define P_BYTES    (P_ELEMS * 2u)               // 16,515,072
#define PT_BYTES   (8u * 256u * 1024u * 2u)     //  4,194,304  p_input^T bf16 [B][D][KPAD]
#define RNN_BYTES  (8000u * 512u * 4u)          // 16,384,000  rnn_in f32 [N*B][2D]
#define XW_BYTES   (8000u * 384u * 4u)          // 12,288,000  xw f32 [N*B][3H]

typedef __bf16 bf16_t;
typedef __attribute__((ext_vector_type(16))) __bf16 v16bf;
typedef __attribute__((ext_vector_type(8)))  __bf16 v8bf;
typedef __attribute__((ext_vector_type(8)))  float  v8f;

// ---- CDNA5 WMMA 16x16x32 bf16 (wave32). A: lane(idx)=row, B: lane(idx)=col.
// 16-bit operand K-pattern per ISA 7.12.2: element e holds
//   K = e + 8*sel            (e < 8)
//   K = e + 8 + 8*sel        (e >= 8)
__device__ __forceinline__ v8f wmma_bf16(v16bf a, v16bf b, v8f c) {
  return __builtin_amdgcn_wmma_f32_16x16x32_bf16(false, a, false, b,
                                                 (short)0, c, false, false);
}

// build a bf16 operand fragment from a row-major f32 matrix
// (A: row = m0+idx, B[k][n] = W[n][k] with W row-major -> row = n0+idx)
__device__ __forceinline__ v16bf frag_from_f32(const float* __restrict__ src,
                                               int row, int ld, int k0, int sel) {
  const float* p = src + (size_t)row * ld + k0 + 8 * sel;
  v16bf out;
#pragma unroll
  for (int e = 0; e < 8; ++e) out[e] = (bf16_t)p[e];
#pragma unroll
  for (int e = 0; e < 8; ++e) out[8 + e] = (bf16_t)p[16 + e];
  return out;
}

// same but source already bf16 and 16B-aligned (two b128 loads)
__device__ __forceinline__ v16bf frag_from_bf16(const bf16_t* __restrict__ src,
                                                int row, int ld, int k0, int sel) {
  const bf16_t* p = src + (size_t)row * ld + k0 + 8 * sel;
  v8bf lo = *(const v8bf*)p;
  v8bf hi = *(const v8bf*)(p + 16);
  return __builtin_shufflevector(lo, hi, 0, 1, 2, 3, 4, 5, 6, 7,
                                 8, 9, 10, 11, 12, 13, 14, 15);
}

// ---------------- stage 1: vp = p_input . W0^T  (8000x256 x 256x128) -------
__global__ void __launch_bounds__(256) k_vp(const float* __restrict__ X,
                                            const float* __restrict__ W0,
                                            float* __restrict__ vp) {
  const int tid  = threadIdx.x;
  const int wave = tid >> 5, lane = tid & 31;
  const int idx  = lane & 15, sel = lane >> 4;
  const int m0   = blockIdx.x * 16;       // 500 M-tiles
  const int h0   = wave * 16;             // 8 N-tiles, one per wave
  v8f acc = {0.f, 0.f, 0.f, 0.f, 0.f, 0.f, 0.f, 0.f};
#pragma unroll
  for (int kk = 0; kk < 8; ++kk) {
    v16bf a = frag_from_f32(X,  m0 + idx, DIM, kk * 32, sel);
    v16bf b = frag_from_f32(W0, h0 + idx, DIM, kk * 32, sel);
    acc = wmma_bf16(a, b, acc);
  }
#pragma unroll
  for (int v = 0; v < 8; ++v) {
    int m = v + 8 * sel;
    vp[(size_t)(m0 + m) * HID + h0 + idx] = acc[v];
  }
}

// ---------------- zero fill (for P padding rows) ----------------------------
__global__ void k_zero(unsigned int* __restrict__ p, unsigned int n) {
  unsigned int i = blockIdx.x * blockDim.x + threadIdx.x;
  if (i < n) p[i] = 0u;
}

// ------------- stage 1b: PT[b][d][j] = bf16(p_input[j][b][d]), j-padded -----
__global__ void __launch_bounds__(256) k_transpose(const float* __restrict__ pin,
                                                   bf16_t* __restrict__ PT) {
  const int j = blockIdx.x * 256 + threadIdx.x;   // 0..1023 (grid.x = 4)
  const int d = blockIdx.y;                       // 0..255
  const int b = blockIdx.z;                       // 0..7
  float v = (j < N_SEQ) ? pin[((size_t)j * BATCH + b) * DIM + d] : 0.f;
  PT[((size_t)b * DIM + d) * KPAD + j] = (bf16_t)v;
}

// ---------------- stage 2: scores + softmax -> bf16 probs -------------------
// one block per (query i, batch b). Score row (1024 f32) lives in LDS.
__global__ void __launch_bounds__(256) k_scores(const float* __restrict__ vp,
                                                const float* __restrict__ vs,
                                                const unsigned char* __restrict__ mask,
                                                bf16_t* __restrict__ P) {
  const int i = blockIdx.x, b = blockIdx.y;
  const int tid = threadIdx.x;
  __shared__ float vpi[HID];
  __shared__ float vsl[HID];
  __shared__ float srow[KPAD];
  __shared__ float red[256];
  if (tid < HID) {
    vpi[tid] = vp[((size_t)i * BATCH + b) * HID + tid];
    vsl[tid] = vs[tid];
  }
  if (tid < KPAD - N_SEQ) srow[N_SEQ + tid] = -1e30f;
  __syncthreads();

  const int w = tid >> 5, lane = tid & 31;
  for (int j = w; j < N_SEQ; j += 8) {           // each wave owns a j-stripe
    const float* pj = vp + ((size_t)j * BATCH + b) * HID + lane * 4;
    float part = 0.f;
#pragma unroll
    for (int q = 0; q < 4; ++q) {
      int h = lane * 4 + q;
      part += vsl[h] * tanhf(pj[q] + vpi[h]);
    }
#pragma unroll
    for (int off = 16; off >= 1; off >>= 1)      // wave32 reduction
      part += __shfl_xor(part, off, 32);
    if (lane == 0)
      srow[j] = mask[(size_t)j * BATCH + b] ? -1e30f : part;
  }
  __syncthreads();

  float m = -1e30f;
#pragma unroll
  for (int q = 0; q < 4; ++q) m = fmaxf(m, srow[tid + q * 256]);
  red[tid] = m;
  __syncthreads();
  for (int s = 128; s > 0; s >>= 1) {
    if (tid < s) red[tid] = fmaxf(red[tid], red[tid + s]);
    __syncthreads();
  }
  const float mx = red[0];
  __syncthreads();

  float ssum = 0.f;
#pragma unroll
  for (int q = 0; q < 4; ++q) {
    float e = __expf(srow[tid + q * 256] - mx);
    srow[tid + q * 256] = e;
    ssum += e;
  }
  red[tid] = ssum;
  __syncthreads();
  for (int s = 128; s > 0; s >>= 1) {
    if (tid < s) red[tid] += red[tid + s];
    __syncthreads();
  }
  const float inv = 1.0f / red[0];
  bf16_t* prow = P + ((size_t)b * NPAD + i) * KPAD;
#pragma unroll
  for (int q = 0; q < 4; ++q) {
    int x = tid + q * 256;
    prow[x] = (bf16_t)(srow[x] * inv);
  }
}

// ---------------- stage 3: att = P . p_input  (per batch, WMMA) -------------
// A = probs bf16 [NPAD x KPAD]; B from PT (bf16, j-contiguous) -> both
// operands are two b128 loads per fragment.
__global__ void __launch_bounds__(256) k_att(const bf16_t* __restrict__ P,
                                             const bf16_t* __restrict__ PT,
                                             float* __restrict__ rnn_in) {
  const int tid  = threadIdx.x;
  const int wave = tid >> 5, lane = tid & 31;
  const int idx  = lane & 15, sel = lane >> 4;
  const int i0   = blockIdx.x * 16;                 // 63 i-tiles (padded)
  const int d0   = (blockIdx.y * 8 + wave) * 16;    // 16 d-tiles
  const int b    = blockIdx.z;
  const bf16_t* Pb  = P  + (size_t)b * NPAD * KPAD;
  const bf16_t* PTb = PT + (size_t)b * DIM * KPAD;
  v8f acc = {0.f, 0.f, 0.f, 0.f, 0.f, 0.f, 0.f, 0.f};
#pragma unroll 4
  for (int kk = 0; kk < KPAD / 32; ++kk) {
    if (kk + 4 < KPAD / 32) {                       // gfx1250 global_prefetch_b8
      __builtin_prefetch(Pb  + (size_t)(i0 + idx) * KPAD + (kk + 4) * 32, 0, 0);
      __builtin_prefetch(PTb + (size_t)(d0 + idx) * KPAD + (kk + 4) * 32, 0, 0);
    }
    v16bf a  = frag_from_bf16(Pb,  i0 + idx, KPAD, kk * 32, sel);
    v16bf bb = frag_from_bf16(PTb, d0 + idx, KPAD, kk * 32, sel);
    acc = wmma_bf16(a, bb, acc);
  }
#pragma unroll
  for (int v = 0; v < 8; ++v) {
    int i = i0 + v + 8 * sel;
    if (i < N_SEQ)
      rnn_in[((size_t)i * BATCH + b) * (2 * DIM) + d0 + idx] = acc[v];
  }
}

// ---------------- stage 3b: rnn_in[:, D:2D] = p_input -----------------------
__global__ void k_concat(const float* __restrict__ pin, float* __restrict__ rnn_in) {
  size_t i = (size_t)blockIdx.x * blockDim.x + threadIdx.x;   // float4 index
  if (i >= (size_t)N_SEQ * BATCH * DIM / 4) return;
  size_t row = i / (DIM / 4);
  int d4 = (int)(i % (DIM / 4));
  const float4* src = (const float4*)pin;
  float4* dst = (float4*)(rnn_in + row * (2 * DIM) + DIM);
  dst[d4] = src[row * (DIM / 4) + d4];
}

// ---------------- stage 4: xw = rnn_in . W_ih^T + b_ih  (WMMA) --------------
__global__ void __launch_bounds__(256) k_xw(const float* __restrict__ X,
                                            const float* __restrict__ W,
                                            const float* __restrict__ bias,
                                            float* __restrict__ out) {
  const int tid  = threadIdx.x;
  const int wave = tid >> 5, lane = tid & 31;
  const int idx  = lane & 15, sel = lane >> 4;
  const int m0   = blockIdx.x * 16;                 // 500 M-tiles
  const int n0   = (blockIdx.y * 8 + wave) * 16;    // 24 N-tiles
  v8f acc = {0.f, 0.f, 0.f, 0.f, 0.f, 0.f, 0.f, 0.f};
#pragma unroll
  for (int kk = 0; kk < 16; ++kk) {
    v16bf a = frag_from_f32(X, m0 + idx, 2 * DIM, kk * 32, sel);
    v16bf b = frag_from_f32(W, n0 + idx, 2 * DIM, kk * 32, sel);
    acc = wmma_bf16(a, b, acc);
  }
  const float bv = bias[n0 + idx];
#pragma unroll
  for (int v = 0; v < 8; ++v) {
    int m = m0 + v + 8 * sel;
    out[(size_t)m * (3 * HID) + n0 + idx] = acc[v] + bv;
  }
}

// ---------------- stage 5: sequential GRU (one block per direction) ---------
// h kept in LDS (f32 + bf16 WMMA-A copy, padded to 16 rows); W_hh held as
// bf16 B-fragments in VGPRs (wave w owns gate tiles 3w..3w+2).
__global__ void __launch_bounds__(256) k_gru(const float* __restrict__ xw_f,
                                             const float* __restrict__ xw_b,
                                             const float* __restrict__ Whh_f,
                                             const float* __restrict__ Whh_b,
                                             const float* __restrict__ bhh_f,
                                             const float* __restrict__ bhh_b,
                                             float* __restrict__ out) {
  const int dir = blockIdx.x;
  const float* xw   = dir ? xw_b  : xw_f;
  const float* Whh  = dir ? Whh_b : Whh_f;
  const float* bhhg = dir ? bhh_b : bhh_f;
  const int tid  = threadIdx.x;
  const int wave = tid >> 5, lane = tid & 31;
  const int idx  = lane & 15, sel = lane >> 4;

  __shared__ float h32[BATCH * HID];
  __shared__ __align__(16) bf16_t hbf[16 * HID];   // rows 8..15 stay zero
  __shared__ float hw[BATCH * 3 * HID];
  __shared__ float bhh[3 * HID];

  for (int x = tid; x < BATCH * HID; x += 256) h32[x] = 0.f;
  for (int x = tid; x < 16 * HID; x += 256) hbf[x] = (bf16_t)0.f;
  for (int x = tid; x < 3 * HID; x += 256) bhh[x] = bhhg[x];
  __syncthreads();

  v16bf Bfrag[3][4];                               // 96 VGPRs of W_hh
#pragma unroll
  for (int g = 0; g < 3; ++g) {
    int n0 = (3 * wave + g) * 16;
#pragma unroll
    for (int kk = 0; kk < 4; ++kk)
      Bfrag[g][kk] = frag_from_f32(Whh, n0 + idx, HID, kk * 32, sel);
  }

  for (int t = 0; t < N_SEQ; ++t) {
    const int i = dir ? (N_SEQ - 1 - t) : t;
    v8f acc0 = {0.f, 0.f, 0.f, 0.f, 0.f, 0.f, 0.f, 0.f};
    v8f acc1 = acc0, acc2 = acc0;
#pragma unroll
    for (int kk = 0; kk < 4; ++kk) {               // hw = h . W_hh^T
      v16bf a = frag_from_bf16(hbf, idx, HID, kk * 32, sel);
      acc0 = wmma_bf16(a, Bfrag[0][kk], acc0);
      acc1 = wmma_bf16(a, Bfrag[1][kk], acc1);
      acc2 = wmma_bf16(a, Bfrag[2][kk], acc2);
    }
    if (sel == 0) {                                 // rows 0..7 = real batch
#pragma unroll
      for (int v = 0; v < 8; ++v) {
        hw[v * (3 * HID) + (3 * wave + 0) * 16 + idx] = acc0[v];
        hw[v * (3 * HID) + (3 * wave + 1) * 16 + idx] = acc1[v];
        hw[v * (3 * HID) + (3 * wave + 2) * 16 + idx] = acc2[v];
      }
    }
    __syncthreads();
#pragma unroll
    for (int q = 0; q < 4; ++q) {                   // gates: 1024 elems
      int x = tid + q * 256;
      int b = x >> 7, hh = x & 127;
      const float* xwp = xw + ((size_t)i * BATCH + b) * (3 * HID);
      float xr = xwp[hh], xz = xwp[HID + hh], xn = xwp[2 * HID + hh];
      float hr = hw[b * (3 * HID) + hh] + bhh[hh];
      float hz = hw[b * (3 * HID) + HID + hh] + bhh[HID + hh];
      float hn = hw[b * (3 * HID) + 2 * HID + hh] + bhh[2 * HID + hh];
      float r = 1.f / (1.f + __expf(-(xr + hr)));
      float z = 1.f / (1.f + __expf(-(xz + hz)));
      float n = tanhf(xn + r * hn);
      float h2 = (1.f - z) * n + z * h32[b * HID + hh];
      h32[b * HID + hh] = h2;
      hbf[b * HID + hh] = (bf16_t)h2;
      out[((size_t)i * BATCH + b) * (2 * HID) + dir * HID + hh] = h2;
    }
    __syncthreads();
  }
}

// ---------------- host side -------------------------------------------------
extern "C" void kernel_launch(void* const* d_in, const int* in_sizes, int n_in,
                              void* d_out, int out_size, void* d_ws, size_t ws_size,
                              hipStream_t stream) {
  (void)in_sizes; (void)n_in; (void)out_size; (void)ws_size;
  const float*         p_input = (const float*)d_in[0];
  const unsigned char* p_mask  = (const unsigned char*)d_in[1];  // jnp.bool_: 1B
  const float*         W0      = (const float*)d_in[2];
  const float*         vs      = (const float*)d_in[3];
  const float*         W_ih_f  = (const float*)d_in[4];
  const float*         W_hh_f  = (const float*)d_in[5];
  const float*         b_ih_f  = (const float*)d_in[6];
  const float*         b_hh_f  = (const float*)d_in[7];
  const float*         W_ih_b  = (const float*)d_in[8];
  const float*         W_hh_b  = (const float*)d_in[9];
  const float*         b_ih_b  = (const float*)d_in[10];
  const float*         b_hh_b  = (const float*)d_in[11];
  float* out = (float*)d_out;

  char* ws = (char*)d_ws;
  float*  vp     = (float*)(ws);
  bf16_t* P      = (bf16_t*)(ws + VP_BYTES);
  bf16_t* PT     = (bf16_t*)(ws + VP_BYTES + P_BYTES);
  float*  rnn_in = (float*)(ws + VP_BYTES + P_BYTES + PT_BYTES);
  float*  xw_f   = (float*)(ws + VP_BYTES + P_BYTES + PT_BYTES + RNN_BYTES);
  float*  xw_b   = (float*)(ws + VP_BYTES + P_BYTES + PT_BYTES + RNN_BYTES + XW_BYTES);

  // 1) vp = p_input.W0^T ; 1b) bf16 transpose of p_input for the att GEMM
  k_vp<<<dim3(500), 256, 0, stream>>>(p_input, W0, vp);
  k_transpose<<<dim3(4, 256, 8), 256, 0, stream>>>(p_input, PT);
  // 2) zero prob buffer (covers padded i rows / j cols), then scores+softmax
  unsigned int pwords = P_ELEMS / 2;
  k_zero<<<dim3((pwords + 255) / 256), 256, 0, stream>>>((unsigned int*)P, pwords);
  k_scores<<<dim3(N_SEQ, BATCH), 256, 0, stream>>>(vp, vs, p_mask, P);
  // 3) att -> rnn_in[:, :D];  p_input -> rnn_in[:, D:]
  k_att<<<dim3(63, 2, 8), 256, 0, stream>>>(P, PT, rnn_in);
  unsigned int c4 = N_SEQ * BATCH * DIM / 4;
  k_concat<<<dim3((c4 + 255) / 256), 256, 0, stream>>>(p_input, rnn_in);
  // 4) input projections for both directions
  k_xw<<<dim3(500, 3), 256, 0, stream>>>(rnn_in, W_ih_f, b_ih_f, xw_f);
  k_xw<<<dim3(500, 3), 256, 0, stream>>>(rnn_in, W_ih_b, b_ih_b, xw_b);
  // 5) sequential bidirectional GRU
  k_gru<<<dim3(2), 256, 0, stream>>>(xw_f, xw_b, W_hh_f, W_hh_b, b_hh_f, b_hh_b, out);
}
